// GRUDCell_78658031059226
// MI455X (gfx1250) — matrix operand order repaired and checked by reading, not compiled, not run on hardware
//
#include <hip/hip_runtime.h>
#include <hip/hip_bf16.h>
#include <math.h>

// ---------------------------------------------------------------------------
// GRU-D cell for MI455X (gfx1250, wave32).
// GEMMs: bf16 v_wmma_f32_16x16x32_bf16, tiles staged in LDS by the Tensor
// Data Mover (TENSOR_LOAD_TO_LDS + s_wait_tensorcnt), fragments via ds_load.
// Epilogues (exp/sigmoid/tanh, decay, gating) fused in fp32.
// ---------------------------------------------------------------------------

typedef __bf16 bf16_t;
typedef __attribute__((ext_vector_type(16))) __bf16 v16bf;
typedef __attribute__((ext_vector_type(8)))  __bf16 v8bf;
typedef __attribute__((ext_vector_type(8)))  float  v8f;
typedef __attribute__((ext_vector_type(4)))  unsigned int v4u;
typedef __attribute__((ext_vector_type(8)))  int v8i;
typedef __attribute__((ext_vector_type(4)))  int v4i;

#define B_SZ 8192
#define D_SZ 256
#define H_SZ 1024
#define KC   1536   // D (x_hat) + D (m) + H (h_decayed)
#define KCH  64     // K-chunk staged in LDS per TDM transfer

static __device__ __forceinline__ bf16_t f2bf(float f) {
    unsigned u = __float_as_uint(f);
    unsigned r = (u + 0x7FFFu + ((u >> 16) & 1u)) >> 16;
    unsigned short s = (unsigned short)r;
    return __builtin_bit_cast(bf16_t, s);
}
static __device__ __forceinline__ float sigm(float x) {
    return 1.0f / (1.0f + __expf(-x));
}
static __device__ __forceinline__ unsigned lds_off(const void* p) {
    // generic LDS pointer: low 32 bits are the LDS byte address
    return (unsigned)(size_t)p;
}

#if __has_builtin(__builtin_amdgcn_tensor_load_to_lds)
#define HAVE_TDM 1
#else
#define HAVE_TDM 0
#endif

#if HAVE_TDM
// Issue one 2D TDM tile load: tileRows x tileK bf16 elements from a row-major
// matrix (row stride ldElems) into LDS at ldsOff, packed tileK*2 B per row.
// D# layout per CDNA5 ISA 8.3/8.4 (group0: count/lds/global/type,
// group1: data_size=2B, tensor dims, tile dims, dim0 stride).
static __device__ __forceinline__ void tdm_load_2d(
    unsigned ldsOff, const bf16_t* gptr, unsigned ldElems,
    unsigned tdim0, unsigned tdim1, unsigned tileK, unsigned tileRows)
{
    unsigned long long ga = (unsigned long long)(size_t)gptr;
    v4u g0;
    g0[0] = 1u;                                              // count=1, user D#
    g0[1] = ldsOff;                                          // lds_addr
    g0[2] = (unsigned)ga;                                    // global_addr[31:0]
    g0[3] = (unsigned)((ga >> 32) & 0x01FFFFFFu) | (2u << 30); // addr[56:32]|type=2
    v8i g1;
    g1[0] = (int)(1u << 16);                                 // data_size=1 (2B)
    g1[1] = (int)((tdim0 & 0xFFFFu) << 16);                  // tensor_dim0[15:0]
    g1[2] = (int)(((tdim0 >> 16) & 0xFFFFu) | ((tdim1 & 0xFFFFu) << 16));
    g1[3] = (int)(((tdim1 >> 16) & 0xFFFFu) | (tileK << 16)); // tile_dim0
    g1[4] = (int)tileRows;                                   // tile_dim1 (dim2=0)
    g1[5] = (int)ldElems;                                    // tensor_dim0_stride
    g1[6] = 0;
    g1[7] = 0;
    v4i g2 = {0, 0, 0, 0};
    v4i g3 = {0, 0, 0, 0};
#if defined(__clang_major__) && (__clang_major__ >= 23)
    v8i g4 = {0, 0, 0, 0, 0, 0, 0, 0};
    __builtin_amdgcn_tensor_load_to_lds(g0, g1, g2, g3, g4, 0);
#else
    __builtin_amdgcn_tensor_load_to_lds(g0, g1, g2, g3, 0);
#endif
}
#endif

// --- weight packing: [W_x (D) | W_m (D) | W_h (H)] rows of length KC, bf16 ---
__global__ void pack_gate_kernel(const float* __restrict__ Wx,
                                 const float* __restrict__ Wm,
                                 const float* __restrict__ Wh,
                                 bf16_t* __restrict__ dst) {
    int idx = blockIdx.x * blockDim.x + threadIdx.x;
    if (idx >= H_SZ * KC) return;
    int row = idx / KC;
    int col = idx - row * KC;
    float v;
    if (col < D_SZ)            v = Wx[row * D_SZ + col];
    else if (col < 2 * D_SZ)   v = Wm[row * D_SZ + (col - D_SZ)];
    else                       v = Wh[row * H_SZ + (col - 2 * D_SZ)];
    dst[idx] = f2bf(v);
}

__global__ void convert_kernel(const float* __restrict__ src,
                               bf16_t* __restrict__ dst, int n) {
    int idx = blockIdx.x * blockDim.x + threadIdx.x;
    if (idx < n) dst[idx] = f2bf(src[idx]);
}

// --- elementwise prep over [B,D]: gamma_x, imputation, x_hat; fill A-matrix --
__global__ void prep_kernel(const float* __restrict__ x,
                            const float* __restrict__ m,
                            const float* __restrict__ delta,
                            const float* __restrict__ xlast,
                            const float* __restrict__ xmean,
                            const float* __restrict__ Wgx,
                            const float* __restrict__ bgx,
                            float* __restrict__ xlast_out,
                            bf16_t* __restrict__ Abig,
                            bf16_t* __restrict__ delta_bf) {
    int idx = blockIdx.x * blockDim.x + threadIdx.x;
    if (idx >= B_SZ * D_SZ) return;
    int b = idx / D_SZ, d = idx - b * D_SZ;
    float mv = m[idx], xv = x[idx], dl = delta[idx];
    float gx   = __expf(-fmaxf(dl * Wgx[d] + bgx[d], 0.0f));
    float xl   = (mv > 0.0f) ? xv : xlast[idx];
    float ximp = gx * xl + (1.0f - gx) * xmean[d];
    float xhat = mv * xv + (1.0f - mv) * ximp;
    xlast_out[idx] = xl;
    size_t base = (size_t)b * KC;
    Abig[base + d]        = f2bf(xhat);
    Abig[base + D_SZ + d] = f2bf(mv);
    delta_bf[idx]         = f2bf(dl);
}

// ---------------------------------------------------------------------------
// bf16 GEMM: C[row,col] = sum_k A[row,k]*W[col,k]. Block=128x128, 8 waves,
// wave tile 32x64 (2x4 WMMA). Per 64-wide K-chunk: wave0/wave1 issue TDM tile
// loads (A,W) into LDS, s_wait_tensorcnt, barrier, then fragments via ds_load.
// WMMA wave32 layouts per ISA 7.12.2:
//   A 16x32: lanes0-15 row=lane, e0-7=K[k..k+7],   e8-15=K[k+16..k+23]
//            lanes16-31 row=lane-16, e0-7=K[k+8..], e8-15=K[k+24..]
//   B 32x16: lanes0-15 col=lane, e0-7=K[k..k+7], e8-15=K[k+16..k+23]; lanes16-31 K[k+8..], K[k+24..]
//   C 16x16: VGPR j -> M=j / j+8 (half-wave), N=lane&15
// ---------------------------------------------------------------------------
template <int MODE>
__global__ __launch_bounds__(256) void gemm_bf16_kernel(
    const bf16_t* __restrict__ A, int lda,
    const bf16_t* __restrict__ W, int ldw,
    int K,
    const float* __restrict__ bias,
    const float* __restrict__ aux0,
    const float* __restrict__ aux1,
    const float* __restrict__ aux2,
    float* __restrict__ out0,
    bf16_t* __restrict__ outb)
{
    __shared__ __align__(16) bf16_t Atile[128 * KCH];   // 16 KB
    __shared__ __align__(16) bf16_t Wtile[128 * KCH];   // 16 KB

    const int lane = threadIdx.x & 31;
    const int wave = threadIdx.x >> 5;
    const int lh   = lane & 15;
    const int sel  = lane >> 4;            // half-wave select
    const int wm   = wave >> 1;            // 0..3 -> 32-row strip
    const int wn   = wave & 1;             // 0..1 -> 64-col strip
    const int mBase = blockIdx.y * 128 + wm * 32;
    const int nBase = blockIdx.x * 128 + wn * 64;

    v8f acc[2][4];
    #pragma unroll
    for (int i = 0; i < 2; ++i)
        #pragma unroll
        for (int j = 0; j < 4; ++j)
            #pragma unroll
            for (int e = 0; e < 8; ++e) acc[i][j][e] = 0.0f;

    const unsigned aOff = lds_off(&Atile[0]);
    const unsigned wOff = lds_off(&Wtile[0]);

    for (int k0 = 0; k0 < K; k0 += KCH) {
        __syncthreads();    // previous chunk's LDS reads done before overwrite
#if HAVE_TDM
        if (wave == 0) {
            tdm_load_2d(aOff, A + (size_t)(blockIdx.y * 128) * lda + k0,
                        (unsigned)lda, (unsigned)lda, 1u << 20, KCH, 128);
            __builtin_amdgcn_s_wait_tensorcnt(0);
        } else if (wave == 1) {
            tdm_load_2d(wOff, W + (size_t)(blockIdx.x * 128) * ldw + k0,
                        (unsigned)ldw, (unsigned)ldw, 1u << 20, KCH, 128);
            __builtin_amdgcn_s_wait_tensorcnt(0);
        }
#else
        // fallback: cooperative synchronous staging
        for (int t = threadIdx.x; t < 128 * KCH / 8; t += 256) {
            int r = t / (KCH / 8), c8 = t % (KCH / 8);
            ((v8bf*)Atile)[t] =
                *(const v8bf*)(A + (size_t)(blockIdx.y * 128 + r) * lda + k0 + c8 * 8);
            ((v8bf*)Wtile)[t] =
                *(const v8bf*)(W + (size_t)(blockIdx.x * 128 + r) * ldw + k0 + c8 * 8);
        }
#endif
        __syncthreads();

        #pragma unroll
        for (int kk = 0; kk < KCH; kk += 32) {
            v16bf afr[2], bfr[4];
            {
                const v8bf* p0 = (const v8bf*)&Atile[(wm * 32 + lh) * KCH + kk + sel * 8];
                const v8bf* p1 = (const v8bf*)&Atile[(wm * 32 + 16 + lh) * KCH + kk + sel * 8];
                v8bf a0 = p0[0], a1 = p0[2];   // K[..+7], K[+16..+23]
                v8bf c0 = p1[0], c1 = p1[2];
                afr[0] = __builtin_shufflevector(a0, a1, 0,1,2,3,4,5,6,7,8,9,10,11,12,13,14,15);
                afr[1] = __builtin_shufflevector(c0, c1, 0,1,2,3,4,5,6,7,8,9,10,11,12,13,14,15);
            }
            #pragma unroll
            for (int j = 0; j < 4; ++j) {
                const v8bf* p = (const v8bf*)&Wtile[(wn * 64 + j * 16 + lh) * KCH + kk + sel * 8];
                v8bf b0 = p[0], b1 = p[2];     // K[..+7], K[+16..+23]
                bfr[j] = __builtin_shufflevector(b0, b1, 0,1,2,3,4,5,6,7,8,9,10,11,12,13,14,15);
            }
            #pragma unroll
            for (int mi = 0; mi < 2; ++mi)
                #pragma unroll
                for (int ni = 0; ni < 4; ++ni)
                    acc[mi][ni] = __builtin_amdgcn_wmma_f32_16x16x32_bf16(
                        false, afr[mi], false, bfr[ni],
                        (short)0, acc[mi][ni], false, false);
        }
    }

    // ---- fused epilogues ----
    #pragma unroll
    for (int mi = 0; mi < 2; ++mi) {
        #pragma unroll
        for (int ni = 0; ni < 4; ++ni) {
            int col = nBase + ni * 16 + lh;
            #pragma unroll
            for (int j = 0; j < 8; ++j) {
                int row = mBase + mi * 16 + sel * 8 + j;
                float pre = acc[mi][ni][j];
                if (MODE == 0) {           // gamma_h -> h_decayed
                    float p  = pre + bias[col];
                    float gh = __expf(-fmaxf(p, 0.0f));
                    size_t id = (size_t)row * H_SZ + col;
                    float hd = gh * aux0[id];          // aux0 = h
                    out0[id] = hd;                     // h_dec fp32
                    outb[(size_t)row * KC + 2 * D_SZ + col] = f2bf(hd);
                } else if (MODE == 1) {    // z (col<H) | r (col>=H)
                    if (col < H_SZ) {
                        size_t id = (size_t)row * H_SZ + col;
                        out0[id] = sigm(pre + bias[col]);          // z
                    } else {
                        int nn = col - H_SZ;
                        size_t id = (size_t)row * H_SZ + nn;
                        float rr = sigm(pre + aux0[nn]);           // aux0 = b_mr
                        outb[id] = f2bf(rr * aux1[id]);            // r*h_dec bf16
                    }
                } else if (MODE == 2) {    // h_tilde partial (x_hat,m terms)+b
                    size_t id = (size_t)row * H_SZ + col;
                    out0[id] = pre + bias[col];
                } else {                   // MODE 3: final combine
                    size_t id = (size_t)row * H_SZ + col;
                    float ht = tanhf(pre + aux0[id]);  // aux0 = t_partial
                    float z  = aux1[id];
                    float hd = aux2[id];
                    out0[id] = (1.0f - z) * hd + z * ht;
                }
            }
        }
    }
}

extern "C" void kernel_launch(void* const* d_in, const int* in_sizes, int n_in,
                              void* d_out, int out_size, void* d_ws, size_t ws_size,
                              hipStream_t stream) {
    const float* x     = (const float*)d_in[0];
    const float* m     = (const float*)d_in[1];
    const float* delta = (const float*)d_in[2];
    const float* h     = (const float*)d_in[3];
    const float* xlast = (const float*)d_in[4];
    const float* xmean = (const float*)d_in[5];
    const float* Wgx   = (const float*)d_in[6];
    const float* bgx   = (const float*)d_in[7];
    const float* Wgh   = (const float*)d_in[8];
    const float* bgh   = (const float*)d_in[9];
    const float* Wxz   = (const float*)d_in[10];
    const float* Whz   = (const float*)d_in[11];
    const float* Wmz   = (const float*)d_in[12];
    const float* bmz   = (const float*)d_in[13];
    const float* Wxr   = (const float*)d_in[14];
    const float* Whr   = (const float*)d_in[15];
    const float* Wmr   = (const float*)d_in[16];
    const float* bmr   = (const float*)d_in[17];
    const float* Wxh   = (const float*)d_in[18];
    const float* Whh   = (const float*)d_in[19];
    const float* Wmh   = (const float*)d_in[20];
    const float* bmh   = (const float*)d_in[21];

    float* h_new     = (float*)d_out;                         // [B,H]
    float* xlast_out = (float*)d_out + (size_t)B_SZ * H_SZ;   // [B,D]

    // workspace carve-up (≈157 MB)
    char* ws = (char*)d_ws;
    size_t off = 0;
    auto carve = [&](size_t bytes) {
        void* p = ws + off;
        off = (off + bytes + 255) & ~(size_t)255;
        return p;
    };
    bf16_t* Wzr   = (bf16_t*)carve((size_t)2 * H_SZ * KC * 2);   // [2H,KC] z|r
    bf16_t* Whc   = (bf16_t*)carve((size_t)H_SZ * KC * 2);       // [H,KC]  h gate
    bf16_t* Wghb  = (bf16_t*)carve((size_t)H_SZ * D_SZ * 2);     // [H,D]
    bf16_t* Abig  = (bf16_t*)carve((size_t)B_SZ * KC * 2);       // [B,KC]
    bf16_t* dltb  = (bf16_t*)carve((size_t)B_SZ * D_SZ * 2);     // [B,D]
    float*  hdec  = (float*)carve((size_t)B_SZ * H_SZ * 4);      // [B,H]
    float*  zbuf  = (float*)carve((size_t)B_SZ * H_SZ * 4);      // [B,H]
    float*  tpart = (float*)carve((size_t)B_SZ * H_SZ * 4);      // [B,H]
    bf16_t* rhb   = (bf16_t*)carve((size_t)B_SZ * H_SZ * 2);     // [B,H]
    (void)ws_size; (void)in_sizes; (void)n_in; (void)out_size;

    const int T = 256;
    int nGate = H_SZ * KC;
    pack_gate_kernel<<<(nGate + T - 1) / T, T, 0, stream>>>(Wxz, Wmz, Whz, Wzr);
    pack_gate_kernel<<<(nGate + T - 1) / T, T, 0, stream>>>(Wxr, Wmr, Whr, Wzr + (size_t)H_SZ * KC);
    pack_gate_kernel<<<(nGate + T - 1) / T, T, 0, stream>>>(Wxh, Wmh, Whh, Whc);
    convert_kernel<<<(H_SZ * D_SZ + T - 1) / T, T, 0, stream>>>(Wgh, Wghb, H_SZ * D_SZ);

    prep_kernel<<<(B_SZ * D_SZ + T - 1) / T, T, 0, stream>>>(
        x, m, delta, xlast, xmean, Wgx, bgx, xlast_out, Abig, dltb);

    dim3 blk(256);
    dim3 gridH(H_SZ / 128, B_SZ / 128);      // N=1024
    dim3 gridZR(2 * H_SZ / 128, B_SZ / 128); // N=2048

    // gamma_h GEMM + h_decayed epilogue (also fills A-matrix h_dec slice)
    gemm_bf16_kernel<0><<<gridH, blk, 0, stream>>>(
        dltb, D_SZ, Wghb, D_SZ, D_SZ, bgh, h, nullptr, nullptr, hdec, Abig);

    // z and r in one K=1536 GEMM over [x_hat|m|h_dec]
    gemm_bf16_kernel<1><<<gridZR, blk, 0, stream>>>(
        Abig, KC, Wzr, KC, KC, bmz, bmr, hdec, nullptr, zbuf, rhb);

    // h_tilde partial: [x_hat|m] @ [W_xh|W_mh]^T + b_mh  (K=512)
    gemm_bf16_kernel<2><<<gridH, blk, 0, stream>>>(
        Abig, KC, Whc, KC, 2 * D_SZ, bmh, nullptr, nullptr, nullptr, tpart, nullptr);

    // (r*h_dec) @ W_hh^T, fused tanh + gate combine -> h_new  (K=1024)
    gemm_bf16_kernel<3><<<gridH, blk, 0, stream>>>(
        rhb, H_SZ, Whc + 2 * D_SZ, KC, H_SZ, nullptr, tpart, zbuf, hdec, h_new, nullptr);
}
